// DIGRAC_node_clustering_50491635532096
// MI455X (gfx1250) — matrix operand order repaired and checked by reading, not compiled, not run on hardware
//
#include <hip/hip_runtime.h>

typedef __attribute__((ext_vector_type(16))) _Float16 v16h;
typedef __attribute__((ext_vector_type(8)))  _Float16 v8h;
typedef __attribute__((ext_vector_type(8)))  float    v8f;

#define K_NODES 100000
#define K_FEAT  128
#define K_HID   32
#define K_CLASS 5
#define K_EDGES 1600000
#define K_FILL  0.5f
#define K_TILES (K_NODES / 16)     // 6250 M-tiles of 16 rows

union AFrag { v16h v; _Float16 h[16]; v8h half8[2]; };
union CFrag { v8f  v; float    f[8];  };

// ---------------------------------------------------------------------------
// Dense embedding: x_s = relu(X@Ws0)@Ws1 ; x_t = relu(X@Wt0)@Wt1
//
// Persistent kernel, grid-stride over 16-row M-tiles. The 20 WMMA weight
// fragments (4 k-tiles x {s0,s1,t0,t1} for stage 1, 4 for stage 2) are built
// ONCE per block, converted to f16, and stored in LDS already swizzled into
// the per-lane B-fragment register layout (32B/lane/fragment). Each tile
// then needs only 16 global_load_b128 for features (the real HBM stream:
// 51.2MB total, ~2.2us at 23.3 TB/s), 2 ds_load_b128 per fragment, and 20
// v_wmma_f32_16x16x32_f16 with f32 accumulation.
// ---------------------------------------------------------------------------
__global__ __launch_bounds__(256) void k_embed(
    const float* __restrict__ feat,
    const float* __restrict__ ws0, const float* __restrict__ ws1,
    const float* __restrict__ wt0, const float* __restrict__ wt1,
    float* __restrict__ xs, float* __restrict__ xt)
{
  // 20 fragments x 32 lanes x 16 f16 = 20KB
  __shared__ _Float16 ldsb[20][32][16];
  // relu intermediate, per wave, per branch: 16KB
  __shared__ _Float16 hbuf[2][8][16][K_HID];

  const int wave = threadIdx.x >> 5;
  const int lane = threadIdx.x & 31;
  const int m   = lane & 15;       // A-frag row id
  const int hi  = lane >> 4;       // half-wave selector
  const int nn  = lane & 15;       // B-frag / C-frag column id
  const int hiK = hi << 4;         // unused; B-frag K offset is hi << 3

  // ---- cooperative weight-fragment build (once per block) ----
#pragma unroll
  for (int rep = 0; rep < 3; ++rep) {
    const int id = wave + rep * 8;
    if (id < 20) {
      AFrag b;
      if (id < 16) {               // stage 1: k-tile t, {bs0,bs1,bt0,bt1}
        const int t = id >> 2;
        const int which = id & 3;
        const float* w0 = (which >= 2) ? wt0 : ws0;
        const int ncol0 = (which & 1) ? 16 : 0;
#pragma unroll
        for (int v = 0; v < 8; ++v) {
          const int k = t * 32 + (hi << 3) + 2 * v + ((v >= 4) ? 8 : 0);
          const float* q = w0 + (size_t)k * K_HID + ncol0 + nn;
          b.h[2 * v]     = (_Float16)q[0];
          b.h[2 * v + 1] = (_Float16)q[K_HID];
        }
      } else {                     // stage 2: {s:b0,s:b1,t:b0,t:b1}
        const int id2 = id - 16;
        const float* w1 = (id2 >= 2) ? wt1 : ws1;
        const int ncol0 = (id2 & 1) ? 16 : 0;
#pragma unroll
        for (int v = 0; v < 8; ++v) {
          const int k = (hi << 3) + 2 * v + ((v >= 4) ? 8 : 0);
          const float* q = w1 + (size_t)k * K_HID + ncol0 + nn;
          b.h[2 * v]     = (_Float16)q[0];
          b.h[2 * v + 1] = (_Float16)q[K_HID];
        }
      }
      *(v8h*)&ldsb[id][lane][0] = b.half8[0];
      *(v8h*)&ldsb[id][lane][8] = b.half8[1];
    }
  }
  __syncthreads();

  // ---- persistent tile loop ----
  const int wid0    = blockIdx.x * 8 + wave;
  const int tstride = gridDim.x * 8;
  for (int tile = wid0; tile < K_TILES; tile += tstride) {
    const int row0 = tile * 16;

    CFrag acc[4];
#pragma unroll
    for (int w = 0; w < 4; ++w)
#pragma unroll
      for (int i = 0; i < 8; ++i) acc[w].f[i] = 0.f;

#pragma unroll
    for (int t = 0; t < 4; ++t) {
      // A-frag: lane m needs K runs [hi*8, hi*8+8) and [16+hi*8, ...)
      // -> four global_load_b128 + packed f32->f16 converts.
      const float* base = feat + (size_t)(row0 + m) * K_FEAT + t * 32 + (hi << 3);
      const float4 f0 = ((const float4*)base)[0];
      const float4 f1 = ((const float4*)base)[1];
      const float4 f2 = ((const float4*)(base + 16))[0];
      const float4 f3 = ((const float4*)(base + 16))[1];
      AFrag a;
      a.h[0]  = (_Float16)f0.x; a.h[1]  = (_Float16)f0.y;
      a.h[2]  = (_Float16)f0.z; a.h[3]  = (_Float16)f0.w;
      a.h[4]  = (_Float16)f1.x; a.h[5]  = (_Float16)f1.y;
      a.h[6]  = (_Float16)f1.z; a.h[7]  = (_Float16)f1.w;
      a.h[8]  = (_Float16)f2.x; a.h[9]  = (_Float16)f2.y;
      a.h[10] = (_Float16)f2.z; a.h[11] = (_Float16)f2.w;
      a.h[12] = (_Float16)f3.x; a.h[13] = (_Float16)f3.y;
      a.h[14] = (_Float16)f3.z; a.h[15] = (_Float16)f3.w;

#pragma unroll
      for (int which = 0; which < 4; ++which) {
        AFrag b;
        b.half8[0] = *(const v8h*)&ldsb[t * 4 + which][lane][0];
        b.half8[1] = *(const v8h*)&ldsb[t * 4 + which][lane][8];
        acc[which].v = __builtin_amdgcn_wmma_f32_16x16x32_f16(
            false, a.v, false, b.v, (short)0, acc[which].v, false, false);
      }
    }

    // ReLU + stage to LDS (per-wave region: in-order DS pipe, no barrier)
#pragma unroll
    for (int v = 0; v < 8; ++v) {
      const int mm = v + (hi << 3);
      hbuf[0][wave][mm][nn]      = (_Float16)fmaxf(acc[0].f[v], 0.f);
      hbuf[0][wave][mm][16 + nn] = (_Float16)fmaxf(acc[1].f[v], 0.f);
      hbuf[1][wave][mm][nn]      = (_Float16)fmaxf(acc[2].f[v], 0.f);
      hbuf[1][wave][mm][16 + nn] = (_Float16)fmaxf(acc[3].f[v], 0.f);
    }

    // Second GEMM per branch: [16x32] @ [32x32] = one K=32 WMMA per N-tile
#pragma unroll
    for (int br = 0; br < 2; ++br) {
      float* out = br ? xt : xs;
      AFrag a;
      a.half8[0] = *(const v8h*)&hbuf[br][wave][m][hi << 3];
      a.half8[1] = *(const v8h*)&hbuf[br][wave][m][16 + (hi << 3)];
      AFrag b0, b1;
      b0.half8[0] = *(const v8h*)&ldsb[16 + 2 * br][lane][0];
      b0.half8[1] = *(const v8h*)&ldsb[16 + 2 * br][lane][8];
      b1.half8[0] = *(const v8h*)&ldsb[17 + 2 * br][lane][0];
      b1.half8[1] = *(const v8h*)&ldsb[17 + 2 * br][lane][8];
      CFrag o0, o1;
#pragma unroll
      for (int i = 0; i < 8; ++i) { o0.f[i] = 0.f; o1.f[i] = 0.f; }
      o0.v = __builtin_amdgcn_wmma_f32_16x16x32_f16(false, a.v, false, b0.v, (short)0, o0.v, false, false);
      o1.v = __builtin_amdgcn_wmma_f32_16x16x32_f16(false, a.v, false, b1.v, (short)0, o1.v, false, false);
#pragma unroll
      for (int v = 0; v < 8; ++v) {
        const int mm = v + (hi << 3);
        out[(size_t)(row0 + mm) * K_HID + nn]      = o0.f[v];
        out[(size_t)(row0 + mm) * K_HID + 16 + nn] = o1.f[v];
      }
    }
  }
}

// -------- Degree / edge-weight normalization (self-loops of 0.5 included) ---
__global__ __launch_bounds__(256) void k_deg_init(float* inv_s, float* inv_t) {
  const int i = blockIdx.x * blockDim.x + threadIdx.x;
  if (i < K_NODES) { inv_s[i] = K_FILL; inv_t[i] = K_FILL; }
}

__global__ __launch_bounds__(256) void k_deg_edges(
    const int* __restrict__ rows, const int* __restrict__ cols,
    const float* __restrict__ w, float* deg_s, float* deg_t) {
  const int e = blockIdx.x * blockDim.x + threadIdx.x;
  if (e < K_EDGES) {
    atomicAdd(deg_s + rows[e], w[e]);
    atomicAdd(deg_t + cols[e], w[e]);
  }
}

__global__ __launch_bounds__(256) void k_deg_invert(float* inv_s, float* inv_t) {
  const int i = blockIdx.x * blockDim.x + threadIdx.x;
  if (i < K_NODES) {
    float d = inv_s[i]; inv_s[i] = d > 0.f ? 1.f / d : 0.f;
    d = inv_t[i];       inv_t[i] = d > 0.f ? 1.f / d : 0.f;
  }
}

__global__ __launch_bounds__(256) void k_edge_norm(
    const int* __restrict__ rows, const int* __restrict__ cols,
    const float* __restrict__ w, const float* __restrict__ inv_s,
    const float* __restrict__ inv_t, float* w_es, float* w_et) {
  const int e = blockIdx.x * blockDim.x + threadIdx.x;
  if (e < K_EDGES) {
    const float we = w[e];
    w_es[e] = inv_s[rows[e]] * we;   // A_s row-normalized
    w_et[e] = inv_t[cols[e]] * we;   // A_t row-normalized
  }
}

// -------- feat = w_hop[0] * x  (float4-wide) --------------------------------
__global__ __launch_bounds__(256) void k_feat_init(
    const float4* __restrict__ xs, const float4* __restrict__ xt,
    const float* __restrict__ wsh, const float* __restrict__ wth,
    float4* fs, float4* ft) {
  const int idx = blockIdx.x * blockDim.x + threadIdx.x;
  if (idx < (K_NODES * K_HID) / 4) {
    const float a = wsh[0], b = wth[0];
    float4 x = xs[idx];
    fs[idx] = make_float4(a * x.x, a * x.y, a * x.z, a * x.w);
    x = xt[idx];
    ft[idx] = make_float4(b * x.x, b * x.y, b * x.z, b * x.w);
  }
}

// -------- One scatter hop: next[dst] += wn[e] * curr[src] -------------------
// One wave per edge; lanes cover the 32 feature channels (coalesced gather
// and coalesced f32 atomics; curr/next arrays are 12.8MB each and live in
// the 192MB L2, so this phase is bounded by L2 atomic throughput, not HBM).
__global__ __launch_bounds__(256) void k_scatter(
    const int* __restrict__ dst, const int* __restrict__ src,
    const float* __restrict__ wn, const float* __restrict__ curr,
    float* next) {
  const int idx = blockIdx.x * blockDim.x + threadIdx.x;
  const int e = idx >> 5;
  const int f = idx & 31;
  if (e < K_EDGES) {
    atomicAdd(next + (size_t)dst[e] * K_HID + f,
              wn[e] * curr[(size_t)src[e] * K_HID + f]);
  }
}

// -------- Self-loop add + hop accumulation (float4-wide, stream-ordered) ----
__global__ __launch_bounds__(256) void k_finalize(
    const float4* __restrict__ curr, const float* __restrict__ inv,
    float4* next, float4* featAcc, const float* __restrict__ whop, int h) {
  const int idx = blockIdx.x * blockDim.x + threadIdx.x;
  if (idx < (K_NODES * K_HID) / 4) {
    const int node = idx >> 3;                 // 8 float4 per node row
    const float sl = inv[node] * K_FILL;       // self-loop weight
    const float wh = whop[h];
    const float4 c = curr[idx];
    float4 nx = next[idx];
    nx.x += sl * c.x; nx.y += sl * c.y; nx.z += sl * c.z; nx.w += sl * c.w;
    next[idx] = nx;                            // becomes curr for next hop
    float4 fa = featAcc[idx];
    fa.x += wh * nx.x; fa.y += wh * nx.y; fa.z += wh * nx.z; fa.w += wh * nx.w;
    featAcc[idx] = fa;
  }
}

// -------- Head: logits, argmax, softmax, log_softmax, L2-normalize ----------
__global__ __launch_bounds__(256) void k_head(
    const float* __restrict__ fs, const float* __restrict__ ft,
    const float* __restrict__ Wp, const float* __restrict__ bias,
    float* __restrict__ out) {
  const int i = blockIdx.x * blockDim.x + threadIdx.x;
  if (i >= K_NODES) return;
  float z[2 * K_HID];
  {
    const float4* ps = (const float4*)(fs + (size_t)i * K_HID);
    const float4* pt = (const float4*)(ft + (size_t)i * K_HID);
#pragma unroll
    for (int j = 0; j < K_HID / 4; ++j) {
      const float4 a = ps[j];
      z[4 * j + 0] = a.x; z[4 * j + 1] = a.y; z[4 * j + 2] = a.z; z[4 * j + 3] = a.w;
      const float4 b = pt[j];
      z[K_HID + 4 * j + 0] = b.x; z[K_HID + 4 * j + 1] = b.y;
      z[K_HID + 4 * j + 2] = b.z; z[K_HID + 4 * j + 3] = b.w;
    }
  }
  float logits[K_CLASS];
#pragma unroll
  for (int c = 0; c < K_CLASS; ++c) logits[c] = bias[c];
#pragma unroll
  for (int k = 0; k < 2 * K_HID; ++k) {
    const float zk = z[k];
#pragma unroll
    for (int c = 0; c < K_CLASS; ++c) logits[c] += zk * Wp[k * K_CLASS + c];
  }
  int best = 0;
  float mx = logits[0];
#pragma unroll
  for (int c = 1; c < K_CLASS; ++c) if (logits[c] > mx) { mx = logits[c]; best = c; }
  float e[K_CLASS], s = 0.f;
#pragma unroll
  for (int c = 0; c < K_CLASS; ++c) { e[c] = __expf(logits[c] - mx); s += e[c]; }
  const float invp = 1.f / s;
  const float lse  = __logf(s);
  float nrm = 0.f;
#pragma unroll
  for (int k = 0; k < 2 * K_HID; ++k) nrm += z[k] * z[k];
  const float rn = 1.f / fmaxf(sqrtf(nrm), 1e-12f);

  float* zn = out;                                     // [N, 64]
  float* lp = out + (size_t)K_NODES * 2 * K_HID;       // [N, 5]
  float* pd = lp + (size_t)K_NODES * K_CLASS;          // [N]
  float* pb = pd + (size_t)K_NODES;                    // [N, 5]
  {
    float4* pz = (float4*)(zn + (size_t)i * 2 * K_HID);
#pragma unroll
    for (int j = 0; j < (2 * K_HID) / 4; ++j)
      pz[j] = make_float4(z[4 * j] * rn, z[4 * j + 1] * rn,
                          z[4 * j + 2] * rn, z[4 * j + 3] * rn);
  }
#pragma unroll
  for (int c = 0; c < K_CLASS; ++c) {
    lp[(size_t)i * K_CLASS + c] = logits[c] - mx - lse;
    pb[(size_t)i * K_CLASS + c] = e[c] * invp;
  }
  pd[i] = (float)best;
}

extern "C" void kernel_launch(void* const* d_in, const int* in_sizes, int n_in,
                              void* d_out, int out_size, void* d_ws, size_t ws_size,
                              hipStream_t stream) {
  (void)in_sizes; (void)n_in; (void)out_size; (void)ws_size;
  const float* features = (const float*)d_in[0];
  const int*   rows     = (const int*)d_in[1];             // edge_index[0]
  const int*   cols     = ((const int*)d_in[1]) + K_EDGES; // edge_index[1]
  const float* ew       = (const float*)d_in[2];
  const float* w_s0     = (const float*)d_in[3];
  const float* w_s1     = (const float*)d_in[4];
  const float* w_t0     = (const float*)d_in[5];
  const float* w_t1     = (const float*)d_in[6];
  const float* ws_hop   = (const float*)d_in[7];
  const float* wt_hop   = (const float*)d_in[8];
  const float* Wp       = (const float*)d_in[9];
  const float* bias     = (const float*)d_in[10];
  float* out = (float*)d_out;

  // workspace layout (floats); total ~90.4 MB
  float* W    = (float*)d_ws;
  float* xs   = W;                  // [N,32] embeddings / hop-2 next (s)
  float* xt   = W + 3200000;        // [N,32] embeddings / hop-2 next (t)
  float* fs   = W + 6400000;        // feat_s accumulator
  float* ft   = W + 9600000;        // feat_t accumulator
  float* nA   = W + 12800000;       // hop-1 next (s)
  float* nB   = W + 16000000;       // hop-1 next (t)
  float* wes  = W + 19200000;       // normalized edge weights (s)
  float* wet  = W + 20800000;       // normalized edge weights (t)
  float* invs = W + 22400000;       // 1/deg_s
  float* invt = W + 22500000;       // 1/deg_t

  const int v4Blocks = ((K_NODES * K_HID / 4) + 255) / 256; // 3125
  const int evBlocks = (K_EDGES * 32) / 256;                // 200000
  const int nBlocks  = (K_NODES + 255) / 256;               // 391
  const int eBlocks  = (K_EDGES + 255) / 256;               // 6250
  const int gBlocks  = 128;   // persistent: 1024 waves, ~6 tiles each

  // 1) fused dual-branch WMMA embedding (persistent, LDS-resident fragments)
  k_embed<<<gBlocks, 256, 0, stream>>>(features, w_s0, w_s1, w_t0, w_t1, xs, xt);

  // 2) degree normalization (self-loop weight 0.5 folded into init)
  k_deg_init<<<nBlocks, 256, 0, stream>>>(invs, invt);
  k_deg_edges<<<eBlocks, 256, 0, stream>>>(rows, cols, ew, invs, invt);
  k_deg_invert<<<nBlocks, 256, 0, stream>>>(invs, invt);
  k_edge_norm<<<eBlocks, 256, 0, stream>>>(rows, cols, ew, invs, invt, wes, wet);

  // 3) hop 0 term
  k_feat_init<<<v4Blocks, 256, 0, stream>>>((const float4*)xs, (const float4*)xt,
                                            ws_hop, wt_hop, (float4*)fs, (float4*)ft);

  // 4) hop 1
  hipMemsetAsync(nA, 0, (size_t)K_NODES * K_HID * sizeof(float), stream);
  hipMemsetAsync(nB, 0, (size_t)K_NODES * K_HID * sizeof(float), stream);
  k_scatter<<<evBlocks, 256, 0, stream>>>(rows, cols, wes, xs, nA);
  k_scatter<<<evBlocks, 256, 0, stream>>>(cols, rows, wet, xt, nB);
  k_finalize<<<v4Blocks, 256, 0, stream>>>((const float4*)xs, invs, (float4*)nA,
                                           (float4*)fs, ws_hop, 1);
  k_finalize<<<v4Blocks, 256, 0, stream>>>((const float4*)xt, invt, (float4*)nB,
                                           (float4*)ft, wt_hop, 1);

  // 5) hop 2 (reuse xs/xt as next buffers — their contents are consumed above)
  hipMemsetAsync(xs, 0, (size_t)K_NODES * K_HID * sizeof(float), stream);
  hipMemsetAsync(xt, 0, (size_t)K_NODES * K_HID * sizeof(float), stream);
  k_scatter<<<evBlocks, 256, 0, stream>>>(rows, cols, wes, nA, xs);
  k_scatter<<<evBlocks, 256, 0, stream>>>(cols, rows, wet, nB, xt);
  k_finalize<<<v4Blocks, 256, 0, stream>>>((const float4*)nA, invs, (float4*)xs,
                                           (float4*)fs, ws_hop, 2);
  k_finalize<<<v4Blocks, 256, 0, stream>>>((const float4*)nB, invt, (float4*)xt,
                                           (float4*)ft, wt_hop, 2);

  // 6) head: logits / softmax / log_softmax / argmax / L2 normalize
  k_head<<<nBlocks, 256, 0, stream>>>(fs, ft, Wp, bias, out);
}
